// SimpleSpatialTransformer_50620484550919
// MI455X (gfx1250) — matrix operand order repaired and checked by reading, compiled, not run on hardware
//
#include <hip/hip_runtime.h>

// ---------------------------------------------------------------------------
// CDNA5 (gfx1250) spatial transformer, bandwidth-optimal fused pipeline:
//   conv1(7x7,3->8)+relu+pool  [WMMA f16->f32, K-order chosen for aligned LDS runs]
//   conv2(5x5,8->10)+relu+pool [WMMA f16->f32, ds_load_b128 A-fragments]
//   fc(27040->6)               [LDS tree reduction]
//   affine bilinear warp       [gather]
// ---------------------------------------------------------------------------

typedef __attribute__((ext_vector_type(16))) _Float16 v16h;  // align 32
typedef __attribute__((ext_vector_type(8)))  _Float16 v8h;   // align 16
typedef __attribute__((ext_vector_type(4)))  _Float16 v4h;   // align 8
typedef __attribute__((ext_vector_type(8)))  float    v8f;

union FragQ { v16h v; v4h q[4]; };  // built from 4x ds_load_b64
union FragR { v16h v; v8h r[2]; };  // built from 2x ds_load_b128

__device__ __forceinline__ v8f wmma_f32_f16(v16h a, v16h b, v8f c) {
    // 8 args: (neg_a, A, neg_b, B, c_mod, C, reuse_a, reuse_b)
    return __builtin_amdgcn_wmma_f32_16x16x32_f16(false, a, false, b,
                                                  (short)0, c, false, false);
}

// ======================= conv1 + bias + relu + maxpool ======================
// x: [64,224,224,3] f32 -> pool1: [64,109,109,8] f16
// K-order: Kp = kh*32 + kw*4 + ci  (ci padded 3->4, kw padded 7->8), 7 k-steps.
// A-fragment 8-run (Kp0 = kk*32+16rho+8h) => kh==kk, contiguous 16B in LDS at
// byte (rowOff+kk)*1792 + colA*8 + 32rho + 16h  (8B aligned -> 2x ds_load_b64).
__global__ __launch_bounds__(256) void conv1_pool_kernel(
    const float* __restrict__ x, const float* __restrict__ w,
    const float* __restrict__ bias, _Float16* __restrict__ pool1)
{
    __shared__ __align__(16) _Float16 xs[8 * 224 * 4 + 8]; // rows 2r..2r+7, ci pad
    __shared__ __align__(32) _Float16 wf[7 * 32 * 16];     // B frags, WMMA layout

    const int tid = threadIdx.x;
    const int r   = blockIdx.x;   // pooled row 0..108
    const int b   = blockIdx.y;

    const float* xb = x + (size_t)b * (224 * 224 * 3) + (size_t)(2 * r) * (224 * 3);
    __builtin_prefetch(xb + tid * 21, 0, 0);   // global_prefetch_b8 (strip warm-up)

    // stage input strip f32x3 -> f16x4 (4th = 0); 8-byte LDS stores
    for (int i = tid; i < 8 * 224; i += 256) {
        const float* s = xb + i * 3;
        v4h v; v[0] = (_Float16)s[0]; v[1] = (_Float16)s[1];
        v[2] = (_Float16)s[2]; v[3] = (_Float16)0.f;
        *(v4h*)&xs[i * 4] = v;
    }
    if (tid < 8) xs[8 * 224 * 4 + tid] = (_Float16)0.f;    // zero tail pad

    // stage weights pre-swizzled to B-fragment layout:
    //   wf[kk*512 + lane*16 + j]: N=lane&15, Kp = kk*32 + 16*(j>>3) + 8*(lane>>4) + (j&7)
    for (int i = tid; i < 7 * 32 * 16; i += 256) {
        int kk = i >> 9, lane = (i >> 4) & 31, j = i & 15;
        int Kp = kk * 32 + ((j >> 3) << 4) + ((lane >> 4) << 3) + (j & 7);
        int n  = lane & 15;
        int kh = Kp >> 5, t = Kp & 31, kw = t >> 2, ci = t & 3;
        _Float16 v = (_Float16)0.f;
        if (kw < 7 && ci < 3 && n < 8)
            v = (_Float16)w[((kh * 7 + kw) * 3 + ci) * 8 + n];
        wf[i] = v;
    }
    __syncthreads();

    const int wave = tid >> 5;
    const int lane = tid & 31;
    const int h    = lane >> 4;   // lane half
    const int m15  = lane & 15;   // A-row (conv col offset) / output channel
    const float bn = (m15 < 8) ? bias[m15] : 0.f;
    const char* px = (const char*)xs;

    for (int cidx = wave; cidx < 14; cidx += 8) {        // 14 chunks x 8 pooled cols
        const int c0 = cidx * 16;
        int colA = c0 + m15; if (colA > 217) colA = 217; // clamp; masked at store
        const int base = colA * 8 + 16 * h;              // per-lane byte base
        v8f accT = {}; v8f accB = {};
        #pragma unroll
        for (int kk = 0; kk < 7; ++kk) {
            v16h bf = *(const v16h*)&wf[kk * 512 + lane * 16];
            FragQ aT, aB;
            #pragma unroll
            for (int rho = 0; rho < 2; ++rho) {
                const int off = base + kk * 1792 + rho * 32;
                aT.q[2 * rho]     = *(const v4h*)(px + off);
                aT.q[2 * rho + 1] = *(const v4h*)(px + off + 8);
                aB.q[2 * rho]     = *(const v4h*)(px + off + 1792);
                aB.q[2 * rho + 1] = *(const v4h*)(px + off + 1792 + 8);
            }
            accT = wmma_f32_f16(aT.v, bf, accT);
            accB = wmma_f32_f16(aB.v, bf, accB);
        }
        // maxpool 2x2: row-pair max (accT/accB), col-pair max (adjacent VGPRs)
        if (m15 < 8) {
            _Float16* dst = pool1 + (((size_t)b * 109 + r) * 109) * 8;
            #pragma unroll
            for (int p = 0; p < 4; ++p) {
                int pc = cidx * 8 + h * 4 + p;
                if (pc < 109) {
                    float v = fmaxf(fmaxf(accT[2 * p],     accB[2 * p]),
                                    fmaxf(accT[2 * p + 1], accB[2 * p + 1]));
                    dst[(size_t)pc * 8 + m15] = (_Float16)fmaxf(v + bn, 0.f);
                }
            }
        }
    }
}

// ======================= conv2 + bias + relu + maxpool ======================
// pool1: [64,109,109,8] f16 -> pool2: [64,52,52,10] f32
// K-order: Kp = kh*40 + kw*8 + ci (segment 40, 200 -> pad 224, 7 k-steps).
// 8-runs are 16B-aligned in LDS -> single ds_load_b128 each.
__host__ __device__ constexpr int c2_run_off(int Kp0) {
    return (Kp0 >= 200) ? 0 : ((Kp0 / 40) * 1744 + 2 * (Kp0 % 40));
}

__global__ __launch_bounds__(256) void conv2_pool_kernel(
    const _Float16* __restrict__ pool1, const float* __restrict__ w,
    const float* __restrict__ bias, float* __restrict__ pool2)
{
    __shared__ __align__(16) _Float16 xs[6 * 109 * 8];   // rows 2r..2r+5
    __shared__ __align__(32) _Float16 wf[7 * 32 * 16];

    const int tid = threadIdx.x;
    const int r   = blockIdx.x;   // pooled row 0..51
    const int b   = blockIdx.y;

    const _Float16* xb = pool1 + ((size_t)b * 109 + 2 * r) * (109 * 8);
    __builtin_prefetch(xb + tid * 16, 0, 0);

    // stage strip: straight 8-byte copies (both sides 8B aligned)
    for (int i = tid; i < (6 * 109 * 8) / 4; i += 256)
        ((v4h*)xs)[i] = ((const v4h*)xb)[i];

    for (int i = tid; i < 7 * 32 * 16; i += 256) {
        int kk = i >> 9, lane = (i >> 4) & 31, j = i & 15;
        int Kp = kk * 32 + ((j >> 3) << 4) + ((lane >> 4) << 3) + (j & 7);
        int n  = lane & 15;
        _Float16 v = (_Float16)0.f;
        if (Kp < 200 && n < 10) {
            int kh = Kp / 40, t = Kp % 40, kw = t >> 3, ci = t & 7;
            v = (_Float16)w[((kh * 5 + kw) * 8 + ci) * 10 + n];
        }
        wf[i] = v;
    }
    __syncthreads();

    const int wave = tid >> 5;
    const int lane = tid & 31;
    const int h    = lane >> 4;
    const int m15  = lane & 15;
    const float bn = (m15 < 10) ? bias[m15] : 0.f;
    const char* px = (const char*)xs;

    if (wave < 7) {                                      // 7 chunks cover 52 cols
        const int c0 = wave * 16;
        int colA = c0 + m15; if (colA > 104) colA = 104;
        const int base = colA * 16;                      // 16B aligned
        v8f accT = {}; v8f accB = {};
        #pragma unroll
        for (int kk = 0; kk < 7; ++kk) {
            v16h bf = *(const v16h*)&wf[kk * 512 + lane * 16];
            FragR aT, aB;
            #pragma unroll
            for (int rho = 0; rho < 2; ++rho) {
                constexpr_helper:;
                const int Kp0 = kk * 32 + 16 * rho;
                const int o0 = c2_run_off(Kp0);
                const int o1 = c2_run_off(Kp0 + 8);
                const int off = base + o0 + h * (o1 - o0);   // 1 v_mad
                aT.r[rho] = *(const v8h*)(px + off);
                aB.r[rho] = *(const v8h*)(px + off + 1744);
            }
            accT = wmma_f32_f16(aT.v, bf, accT);
            accB = wmma_f32_f16(aB.v, bf, accB);
        }
        if (m15 < 10) {
            float* dst = pool2 + (((size_t)b * 52 + r) * 52) * 10;
            #pragma unroll
            for (int p = 0; p < 4; ++p) {
                int pc = wave * 8 + h * 4 + p;
                if (pc < 52) {
                    float v = fmaxf(fmaxf(accT[2 * p],     accB[2 * p]),
                                    fmaxf(accT[2 * p + 1], accB[2 * p + 1]));
                    dst[(size_t)pc * 10 + m15] = fmaxf(v + bn, 0.f);
                }
            }
        }
    }
}

// ============================ fc: theta = feat@W + b ========================
__global__ __launch_bounds__(256) void fc_kernel(
    const float* __restrict__ feat, const float* __restrict__ fw,
    const float* __restrict__ fb, float* __restrict__ theta)
{
    __shared__ float red[6 * 256];
    const int b = blockIdx.x, tid = threadIdx.x;
    const float* f = feat + (size_t)b * 27040;
    float acc[6] = {0.f, 0.f, 0.f, 0.f, 0.f, 0.f};
    for (int i = tid; i < 27040; i += 256) {
        float v = f[i];
        const float* wr = fw + (size_t)i * 6;
        #pragma unroll
        for (int j = 0; j < 6; ++j) acc[j] += v * wr[j];
    }
    #pragma unroll
    for (int j = 0; j < 6; ++j) red[j * 256 + tid] = acc[j];
    __syncthreads();
    for (int s = 128; s > 0; s >>= 1) {
        if (tid < s) {
            #pragma unroll
            for (int j = 0; j < 6; ++j)
                red[j * 256 + tid] += red[j * 256 + tid + s];
        }
        __syncthreads();
    }
    if (tid < 6) theta[b * 6 + tid] = red[tid * 256] + fb[tid];
}

// ===================== affine bilinear warp (zero fill OOB) =================
__global__ __launch_bounds__(256) void warp_kernel(
    const float* __restrict__ x, const float* __restrict__ theta,
    float* __restrict__ out)
{
    size_t idx = (size_t)blockIdx.x * 256 + threadIdx.x;
    if (idx >= (size_t)64 * 224 * 224) return;
    int b   = (int)(idx / (224 * 224));
    int pix = (int)(idx % (224 * 224));
    int yo = pix / 224, xo = pix % 224;

    const float* th = theta + b * 6;
    float xq = th[0] * (float)xo + th[1] * (float)yo + th[2];
    float yq = th[3] * (float)xo + th[4] * (float)yo + th[5];
    float x0f = floorf(xq), y0f = floorf(yq);
    float dx = xq - x0f, dy = yq - y0f;
    int x0 = (int)x0f, y0 = (int)y0f;

    float w00 = (1.f - dx) * (1.f - dy);
    float w01 = dx * (1.f - dy);
    float w10 = (1.f - dx) * dy;
    float w11 = dx * dy;

    const float* xb = x + (size_t)b * (224 * 224 * 3);
    float* ob = out + idx * 3;
    #pragma unroll
    for (int c = 0; c < 3; ++c) {
        auto rd = [&](int yy, int xx) -> float {
            if (yy < 0 || yy > 223 || xx < 0 || xx > 223) return 0.f;
            return xb[((size_t)yy * 224 + xx) * 3 + c];
        };
        float v00 = rd(y0, x0),     v01 = rd(y0, x0 + 1);
        float v10 = rd(y0 + 1, x0), v11 = rd(y0 + 1, x0 + 1);
        ob[c] = w00 * v00 + w01 * v01 + w10 * v10 + w11 * v11;
    }
}

// ================================ launcher ==================================
extern "C" void kernel_launch(void* const* d_in, const int* in_sizes, int n_in,
                              void* d_out, int out_size, void* d_ws, size_t ws_size,
                              hipStream_t stream) {
    const float* x       = (const float*)d_in[0];
    const float* conv1_w = (const float*)d_in[1];
    const float* conv1_b = (const float*)d_in[2];
    const float* conv2_w = (const float*)d_in[3];
    const float* conv2_b = (const float*)d_in[4];
    const float* fc_w    = (const float*)d_in[5];
    const float* fc_b    = (const float*)d_in[6];
    float* out = (float*)d_out;

    const size_t POOL1_BYTES = (size_t)64 * 109 * 109 * 8 * sizeof(_Float16); // 12.2 MB
    const size_t POOL2_BYTES = (size_t)64 * 52 * 52 * 10 * sizeof(float);     //  6.9 MB
    char* ws = (char*)d_ws;
    _Float16* pool1 = (_Float16*)ws;
    float*    pool2 = (float*)(ws + POOL1_BYTES);
    float*    theta = (float*)(ws + POOL1_BYTES + POOL2_BYTES);

    conv1_pool_kernel<<<dim3(109, 64), 256, 0, stream>>>(x, conv1_w, conv1_b, pool1);
    conv2_pool_kernel<<<dim3(52, 64),  256, 0, stream>>>(pool1, conv2_w, conv2_b, pool2);
    fc_kernel<<<64, 256, 0, stream>>>(pool2, fc_w, fc_b, theta);
    warp_kernel<<<(unsigned)(((size_t)64 * 224 * 224 + 255) / 256), 256, 0, stream>>>(x, theta, out);
}
